// HMM_11312943857913
// MI455X (gfx1250) — matrix operand order repaired and checked by reading, not compiled, not run on hardware
//
#include <hip/hip_runtime.h>
#include <hip/hip_bf16.h>
#include <math.h>

// ---------------------------------------------------------------------------
// HMM log-forward (evidence) via linear-domain semiring matrix scan + WMMA.
//
//   f_{t+1} = (f_t^T A_lin) * e_t         (linear domain, rescaled)
//   => chunk products P_c = prod_t (A_lin diag(ê_t)) computed with
//      v_wmma_f32_16x16x32_f16 on 80x80 (5x5 tiles, K padded to 96),
//      deferred 1/max rescale (one barrier per step), emissions
//      pre-normalized per observation (max folded into the log-scale),
//      then combined sequentially (cheap) with the initial distribution.
// ---------------------------------------------------------------------------

typedef __attribute__((ext_vector_type(16))) _Float16 v16h;
typedef __attribute__((ext_vector_type(8)))  _Float16 v8h;
typedef __attribute__((ext_vector_type(8)))  float    v8f;

#define S_STATES 65
#define SP       80      // states padded to 5x16 (M/N tiles)
#define KP       96      // K padded to 3x32
#define PSTRIDE  104     // LDS row stride in halves (208B = 13*16B, bank friendly)
#define NCHUNK   1024
#define OBS_TILE 256

static __device__ __forceinline__ v8f wmma_step(v16h a, v16h b, v8f c) {
    return __builtin_amdgcn_wmma_f32_16x16x32_f16(false, a, false, b, (short)0, c,
                                                  false, false);
}

// Build an A-operand v16h per the CDNA5 16-bit 16x32 A layout:
// lanes<16 hold K {0..7, 16..23}; lanes>=16 hold K {8..15, 24..31}.
// rp already includes the per-lane +8-half offset for upper lanes.
static __device__ __forceinline__ v16h load_aop(const _Float16* rp) {
    v8h lo = *(const v8h*)(rp);
    v8h hi = *(const v8h*)(rp + 16);
    return __builtin_shufflevector(lo, hi, 0, 1, 2, 3, 4, 5, 6, 7,
                                           8, 9, 10, 11, 12, 13, 14, 15);
}

static __device__ __forceinline__ v8f row_mul(const _Float16* rp,
                                              v16h B0, v16h B1, v16h B2) {
    v8f acc = {};
    acc = wmma_step(load_aop(rp +  0), B0, acc);
    acc = wmma_step(load_aop(rp + 32), B1, acc);
    acc = wmma_step(load_aop(rp + 64), B2, acc);
    return acc;
}

// --------------------------- prep: transition ------------------------------
// AT[n][k] = exp(log_softmax(log_trans, dim=1))[k][n], f16, zero padded to
// [SP=80][KP=96] so it can be loaded directly as WMMA B-operands
// (lane = column n; K {0..7,16..23} per lane with a +8 offset for lanes>=16).
__global__ __launch_bounds__(96) void hmm_prep_trans(
        const float* __restrict__ log_trans, _Float16* __restrict__ AT) {
    __shared__ float lt[S_STATES * S_STATES];
    __shared__ float denom[S_STATES];
    const int tid = threadIdx.x;
    for (int i = tid; i < S_STATES * S_STATES; i += 96) lt[i] = log_trans[i];
    __syncthreads();
    if (tid < S_STATES) {
        float m = -3.4e38f;
        for (int j = 0; j < S_STATES; ++j) m = fmaxf(m, lt[tid * S_STATES + j]);
        float s = 0.f;
        for (int j = 0; j < S_STATES; ++j) s += expf(lt[tid * S_STATES + j] - m);
        denom[tid] = m + logf(s);
    }
    __syncthreads();
    const int n = tid;
    for (int k = 0; k < KP; ++k) {
        float v = 0.f;
        if (n < S_STATES && k < S_STATES) v = expf(lt[k * S_STATES + n] - denom[k]);
        if (n < SP) AT[n * KP + k] = (_Float16)v;
    }
}

// --------------------------- prep: emission --------------------------------
// ElinT[v][s] = exp(log_softmax(log_em, dim=1))[s][v], f32, [V][SP] (pad 0).
__global__ __launch_bounds__(256) void hmm_prep_em(
        const float* __restrict__ log_em, float* __restrict__ ElinT, int V) {
    __shared__ float red[256];
    const int s = blockIdx.x;     // 0..79
    const int tid = threadIdx.x;
    if (s >= S_STATES) {
        for (int v = tid; v < V; v += 256) ElinT[(size_t)v * SP + s] = 0.f;
        return;
    }
    const float* row = log_em + (size_t)s * V;
    float m = -3.4e38f;
    for (int v = tid; v < V; v += 256) m = fmaxf(m, row[v]);
    red[tid] = m; __syncthreads();
    for (int off = 128; off; off >>= 1) {
        if (tid < off) red[tid] = fmaxf(red[tid], red[tid + off]);
        __syncthreads();
    }
    const float bm = red[0]; __syncthreads();
    float sum = 0.f;
    for (int v = tid; v < V; v += 256) sum += expf(row[v] - bm);
    red[tid] = sum; __syncthreads();
    for (int off = 128; off; off >>= 1) {
        if (tid < off) red[tid] += red[tid + off];
        __syncthreads();
    }
    const float dn = bm + logf(red[0]);
    for (int v = tid; v < V; v += 256)
        ElinT[(size_t)v * SP + s] = expf(row[v] - dn);
}

// Normalize each observation's emission column to max 1 so the per-step
// matrix decay is O(1); the exact max goes into lemax[v] (added to the
// chunk log-scale deterministically).
__global__ __launch_bounds__(256) void hmm_prep_colmax(
        float* __restrict__ ElinT, float* __restrict__ lemax, int V) {
    const int v = blockIdx.x * 256 + threadIdx.x;
    if (v >= V) return;
    float m = 0.f;
    for (int s = 0; s < S_STATES; ++s) m = fmaxf(m, ElinT[(size_t)v * SP + s]);
    const float inv = (m > 0.f) ? (1.f / m) : 1.f;
    for (int s = 0; s < S_STATES; ++s) ElinT[(size_t)v * SP + s] *= inv;
    lemax[v] = (m > 0.f) ? logf(m) : 0.f;
}

// --------------------------- phase 1: chunk products -----------------------
// Block = 5 waves (160 threads); wave w owns output tile-column nt = w.
// Deferred rescale: step t writes OUT scaled only by 1/gm_{t-1} (folded into
// the emission scalar), so the f16 write-back does not wait on the cross-wave
// max -> ONE barrier per step. P double-buffered; wmax parity-indexed.
__global__ __launch_bounds__(160) void hmm_chunk(
        const int* __restrict__ obvs, const _Float16* __restrict__ AT,
        const float* __restrict__ ElinT, const float* __restrict__ lemax,
        float* __restrict__ Pc, float* __restrict__ lsc, int steps, int L) {
    const int tid  = threadIdx.x;
    const int lane = tid & 31;
    const int w    = tid >> 5;                // wave id = output tile column
    const int c    = blockIdx.x;

    __shared__ __align__(16) _Float16 P[2][SP * PSTRIDE];
    __shared__ int   obsbuf[OBS_TILE];
    __shared__ float lembuf[OBS_TILE];
    __shared__ float wmax[2][8];

    // zero both buffers (incl. K-pad cols 80..95), identity into buffer 0
    for (int i = tid; i < 2 * SP * PSTRIDE; i += 160)
        ((_Float16*)P)[i] = (_Float16)0.f;
    __syncthreads();
    if (tid < S_STATES) P[0][tid * PSTRIDE + tid] = (_Float16)1.f;

    const int m15  = lane & 15;
    const int kh16 = (lane >= 16) ? 16 : 0;   // unused; B loads use kh8
    const int kh8  = (lane >= 16) ? 8  : 0;   // A-operand K offset / C row offset
    const int ncol = w * 16 + m15;            // this lane's output column

    // Constant transition matrix as B operands (3 K-tiles for column nt=w).
    v16h B0 = load_aop(AT + ncol * KP +  0 + kh8);
    v16h B1 = load_aop(AT + ncol * KP + 32 + kh8);
    v16h B2 = load_aop(AT + ncol * KP + 64 + kh8);

    float inv = 1.f;          // 1/gm of the previous step (deferred)
    float pacc = 1.f;         // running product of gm (flushed to log)
    float logscale = 0.f;
    int rb = 0;               // read-buffer index (toggles each step)

    int limit = steps - c * L;
    if (limit > L) limit = L;
    if (limit < 0) limit = 0;

    for (int base = 0; base < limit; base += OBS_TILE) {
        int tile = limit - base;
        if (tile > OBS_TILE) tile = OBS_TILE;
        __syncthreads();
        for (int i = tid; i < tile; i += 160) {
            const int o = obvs[1 + c * L + base + i];
            obsbuf[i] = o;
            lembuf[i] = lemax[o];
        }
        __syncthreads();

        for (int l = 0; l < tile; ++l) {
            const int   obs    = obsbuf[l];
            const float escale = ElinT[(size_t)obs * SP + ncol] * inv;
            logscale += lembuf[l];

            const _Float16* rbase = &P[rb][m15 * PSTRIDE + kh8];
            v8f a0 = row_mul(rbase + 0 * 16 * PSTRIDE, B0, B1, B2);
            v8f a1 = row_mul(rbase + 1 * 16 * PSTRIDE, B0, B1, B2);
            v8f a2 = row_mul(rbase + 2 * 16 * PSTRIDE, B0, B1, B2);
            v8f a3 = row_mul(rbase + 3 * 16 * PSTRIDE, B0, B1, B2);
            v8f a4 = row_mul(rbase + 4 * 16 * PSTRIDE, B0, B1, B2);

            // emission (pre-scaled by previous 1/gm) + per-lane max
            float m = 0.f;
#pragma unroll
            for (int q = 0; q < 8; ++q) {
                a0[q] *= escale; m = fmaxf(m, a0[q]);
                a1[q] *= escale; m = fmaxf(m, a1[q]);
                a2[q] *= escale; m = fmaxf(m, a2[q]);
                a3[q] *= escale; m = fmaxf(m, a3[q]);
                a4[q] *= escale; m = fmaxf(m, a4[q]);
            }

            // write next P immediately (no dependence on this step's max)
            _Float16* wp = &P[1 - rb][kh8 * PSTRIDE + ncol];
#pragma unroll
            for (int q = 0; q < 8; ++q) {
                wp[(0 * 16 + q) * PSTRIDE] = (_Float16)a0[q];
                wp[(1 * 16 + q) * PSTRIDE] = (_Float16)a1[q];
                wp[(2 * 16 + q) * PSTRIDE] = (_Float16)a2[q];
                wp[(3 * 16 + q) * PSTRIDE] = (_Float16)a3[q];
                wp[(4 * 16 + q) * PSTRIDE] = (_Float16)a4[q];
            }
#pragma unroll
            for (int off = 16; off; off >>= 1)
                m = fmaxf(m, __shfl_xor(m, off, 32));
            if (lane == 0) wmax[rb][w] = m;
            __syncthreads();   // single barrier: covers P writes + wmax

            const float gm = fmaxf(fmaxf(fmaxf(wmax[rb][0], wmax[rb][1]),
                                         fmaxf(wmax[rb][2], wmax[rb][3])),
                                   wmax[rb][4]);
            inv = (gm > 0.f) ? (1.f / gm) : 1.f;
            pacc *= (gm > 0.f) ? gm : 1.f;
            if ((l & 15) == 15) { logscale += logf(pacc); pacc = 1.f; }
            rb ^= 1;
        }
        logscale += logf(pacc);
        pacc = 1.f;
    }

    // emit chunk product (f32, compensating the last deferred 1/gm) + scale
    __syncthreads();
    float* out = Pc + (size_t)c * SP * SP;
    for (int i = tid; i < SP * SP; i += 160) {
        int r = i / SP, col = i % SP;
        out[i] = (float)P[rb][r * PSTRIDE + col] * inv;
    }
    if (tid == 0) lsc[c] = logscale;
}

// --------------------------- phase 2: combine ------------------------------
__global__ __launch_bounds__(128) void hmm_combine(
        const float* __restrict__ Pc, const float* __restrict__ lsc,
        const float* __restrict__ log_trans, float* __restrict__ out, int C) {
    __shared__ float vcur[SP];
    __shared__ float red[128];
    const int tid = threadIdx.x;
    if (tid < SP)
        vcur[tid] = (tid == 0) ? 1.f : ((tid < S_STATES) ? expf(-20.f) : 0.f);
    float total = 0.f;
    __syncthreads();
    for (int c = 0; c < C; ++c) {
        float s = 0.f;
        if (tid < SP) {
            const float* M = Pc + (size_t)c * SP * SP;
            for (int i = 0; i < S_STATES; ++i) s += vcur[i] * M[i * SP + tid];
        }
        red[tid] = (tid < SP) ? s : 0.f;
        __syncthreads();
        for (int off = 64; off; off >>= 1) {
            if (tid < off) red[tid] = fmaxf(red[tid], red[tid + off]);
            __syncthreads();
        }
        float gm  = red[0];
        float inv = (gm > 0.f) ? (1.f / gm) : 1.f;
        total += ((gm > 0.f) ? logf(gm) : 0.f) + lsc[c];
        __syncthreads();
        if (tid < SP) vcur[tid] = s * inv;
        __syncthreads();
    }
    // termination: raw log_trans[:,0], then logsumexp
    float wv = 0.f;
    if (tid < S_STATES) wv = vcur[tid] * expf(log_trans[tid * S_STATES]);
    red[tid] = wv;
    __syncthreads();
    for (int off = 64; off; off >>= 1) {
        if (tid < off) red[tid] += red[tid + off];
        __syncthreads();
    }
    if (tid == 0) out[0] = logf(red[0]) + total;
}

// ---------------------------------------------------------------------------
extern "C" void kernel_launch(void* const* d_in, const int* in_sizes, int n_in,
                              void* d_out, int out_size, void* d_ws, size_t ws_size,
                              hipStream_t stream) {
    const int*   obvs      = (const int*)d_in[0];
    const float* log_trans = (const float*)d_in[1];
    const float* log_em    = (const float*)d_in[2];

    const int T     = in_sizes[0];
    const int V     = in_sizes[2] / S_STATES;
    const int steps = T - 2;
    const int C     = NCHUNK;
    const int L     = (steps + C - 1) / C;

    char* ws = (char*)d_ws;
    size_t off = 0;
    _Float16* AT = (_Float16*)(ws + off);
    off += (size_t)SP * KP * sizeof(_Float16);
    off = (off + 255) & ~(size_t)255;
    float* ElinT = (float*)(ws + off);
    off += (size_t)V * SP * sizeof(float);
    off = (off + 255) & ~(size_t)255;
    float* lemax = (float*)(ws + off);
    off += (size_t)V * sizeof(float);
    off = (off + 255) & ~(size_t)255;
    float* lsc = (float*)(ws + off);
    off += (size_t)C * sizeof(float);
    off = (off + 255) & ~(size_t)255;
    float* Pc = (float*)(ws + off);

    hmm_prep_trans<<<1, 96, 0, stream>>>(log_trans, AT);
    hmm_prep_em<<<SP, 256, 0, stream>>>(log_em, ElinT, V);
    hmm_prep_colmax<<<(V + 255) / 256, 256, 0, stream>>>(ElinT, lemax, V);
    hmm_chunk<<<C, 160, 0, stream>>>(obvs, AT, ElinT, lemax, Pc, lsc, steps, L);
    hmm_combine<<<1, 128, 0, stream>>>(Pc, lsc, log_trans, (float*)d_out, C);
}
